// HierarchicalClassifier_89584427860151
// MI455X (gfx1250) — matrix operand order repaired and checked by reading, not compiled, not run on hardware
//
#include <hip/hip_runtime.h>
#include <hip/hip_bf16.h>

typedef __attribute__((ext_vector_type(16))) __bf16 v16bf;
typedef __attribute__((ext_vector_type(8)))  __bf16 v8bf;
typedef __attribute__((ext_vector_type(8)))  float  v8f;

#define DIMD 2048
#define NTOP 8
#define NCLS 16
#define NOUT 128            // N_TOP * N_CLASS
#define NCOL 144            // 8 top + 128 bottom + 8 zero-pad -> 9 WMMA N-tiles
#define NTILE 9
#define NSTRIP 2            // 16-row strips per wave (M=32 per wave)
#define ROWS_PER_BLOCK 64   // 2 waves x 32 rows
#define THREADS 64

// ---------------------------------------------------------------------------
// Prep: build combined weight matrix W[144][2048] column-major (K contiguous),
// split each fp32 into bf16 hi + bf16 lo planes (near-fp32 precision GEMM).
// col n in [0,8): top_W[n][k].  n in [8,136): bottom_W[t][k][c], t=(n-8)/16,
// c=(n-8)%16.  n in [136,144): zero pad.
// ---------------------------------------------------------------------------
__global__ void hc_prep_weights(const float* __restrict__ topW,
                                const float* __restrict__ botW,
                                __bf16* __restrict__ whi,
                                __bf16* __restrict__ wlo) {
  int idx = blockIdx.x * blockDim.x + threadIdx.x;
  if (idx >= NCOL * DIMD) return;
  int n = idx / DIMD;
  int k = idx - n * DIMD;
  float w = 0.0f;
  if (n < NTOP) {
    w = topW[n * DIMD + k];
  } else if (n < NTOP + NOUT) {
    int nc = n - NTOP;
    int t = nc >> 4, c = nc & 15;
    w = botW[((size_t)t * DIMD + k) * NCLS + c];
  }
  __bf16 hi = (__bf16)w;
  __bf16 lo = (__bf16)(w - (float)hi);
  whi[idx] = hi;
  wlo[idx] = lo;
}

// ---------------------------------------------------------------------------
// Main: fused GEMM (split-bf16 WMMA, fp32 accumulate) + sigmoid/softmax
// epilogue.  Each wave computes two 16-row strips (M=32) x 144 cols, so each
// B fragment pair (hi/lo) is amortized over 6 WMMAs.
// ---------------------------------------------------------------------------
__launch_bounds__(THREADS)
__global__ void hc_main(const float* __restrict__ feat,
                        const __bf16* __restrict__ whi,
                        const __bf16* __restrict__ wlo,
                        const float* __restrict__ top_b,
                        const float* __restrict__ bot_b,
                        float* __restrict__ out) {
  __shared__ float smem[ROWS_PER_BLOCK * NCOL];

  const int tid   = threadIdx.x;
  const int wave  = tid >> 5;
  const int lane  = tid & 31;
  const int laneM = lane & 15;   // row within strip (A/C), column within tile (B/C)
  const int half  = lane >> 4;   // which K/M half of the fragment this lane holds

  const int rowBase = blockIdx.x * ROWS_PER_BLOCK + wave * (NSTRIP * 16);
  const float* arow[NSTRIP];
#pragma unroll
  for (int s = 0; s < NSTRIP; ++s)
    arow[s] = feat + (size_t)(rowBase + s * 16 + laneM) * DIMD + half * 8;

  v8f acc[NSTRIP][NTILE];
#pragma unroll
  for (int s = 0; s < NSTRIP; ++s)
#pragma unroll
    for (int t = 0; t < NTILE; ++t)
      acc[s][t] = v8f{0.f, 0.f, 0.f, 0.f, 0.f, 0.f, 0.f, 0.f};

  // A-fragment staging registers (current chunk) for software pipelining
  float4 acur[NSTRIP][4];
#pragma unroll
  for (int s = 0; s < NSTRIP; ++s) {
    const float4* ap = (const float4*)(arow[s]);
    acur[s][0] = ap[0]; acur[s][1] = ap[1];   // K = half*8 + 0..7
    acur[s][2] = ap[4]; acur[s][3] = ap[5];   // K = 16 + half*8 + 0..7
  }

  for (int kc = 0; kc < DIMD; kc += 32) {
    // ---- convert current A chunk to bf16 hi/lo fragments
    // A fragment (16x32 bf16 layout): lanes 0-15 hold K 0-7 & 16-23,
    // lanes 16-31 hold K 8-15 & 24-31, packed 2 bf16 per VGPR in order.
    v16bf ahi[NSTRIP], alo[NSTRIP];
#pragma unroll
    for (int s = 0; s < NSTRIP; ++s) {
      float av[16] = {acur[s][0].x, acur[s][0].y, acur[s][0].z, acur[s][0].w,
                      acur[s][1].x, acur[s][1].y, acur[s][1].z, acur[s][1].w,
                      acur[s][2].x, acur[s][2].y, acur[s][2].z, acur[s][2].w,
                      acur[s][3].x, acur[s][3].y, acur[s][3].z, acur[s][3].w};
#pragma unroll
      for (int i = 0; i < 16; ++i) {
        __bf16 h = (__bf16)av[i];
        ahi[s][i] = h;
        alo[s][i] = (__bf16)(av[i] - (float)h);
      }
    }

    // ---- prefetch next A chunk (overlaps with the 9-tile WMMA loop)
    if (kc + 32 < DIMD) {
#pragma unroll
      for (int s = 0; s < NSTRIP; ++s) {
        const float4* ap = (const float4*)(arow[s] + kc + 32);
        acur[s][0] = ap[0]; acur[s][1] = ap[1];
        acur[s][2] = ap[4]; acur[s][3] = ap[5];
      }
    }

#pragma unroll
    for (int t = 0; t < NTILE; ++t) {
      // ---- B fragment (32x16 bf16): lane n = column n (K 0-7 & 16-23),
      //      lane 16+n = column n (K 8-15 & 24-31); columns are K-contiguous in ws.
      size_t boff = (size_t)(t * 16 + laneM) * DIMD + kc + half * 8;
      v8bf bh0 = *(const v8bf*)(whi + boff);
      v8bf bh1 = *(const v8bf*)(whi + boff + 16);
      v8bf bl0 = *(const v8bf*)(wlo + boff);
      v8bf bl1 = *(const v8bf*)(wlo + boff + 16);
      v16bf bhi, blo;
#pragma unroll
      for (int i = 0; i < 8; ++i) {
        bhi[i] = bh0[i]; bhi[i + 8] = bh1[i];
        blo[i] = bl0[i]; blo[i + 8] = bl1[i];
      }
      // split-bf16: hi*hi + lo*hi + hi*lo ~ fp32 product, fp32 accumulate.
      // One B pair feeds both row strips (6 WMMAs per 4 B loads).
#pragma unroll
      for (int s = 0; s < NSTRIP; ++s) {
        acc[s][t] = __builtin_amdgcn_wmma_f32_16x16x32_bf16(
            false, ahi[s], false, bhi, (short)0, acc[s][t], false, false);
        acc[s][t] = __builtin_amdgcn_wmma_f32_16x16x32_bf16(
            false, alo[s], false, bhi, (short)0, acc[s][t], false, false);
        acc[s][t] = __builtin_amdgcn_wmma_f32_16x16x32_bf16(
            false, ahi[s], false, blo, (short)0, acc[s][t], false, false);
      }
    }
  }

  // ---- Spill C fragments to LDS (C layout: VGPR r -> row r (lanes 0-15) or
  //      row 8+r (lanes 16-31), lane%16 -> column).
  float* wbase = smem + wave * (NSTRIP * 16) * NCOL;
#pragma unroll
  for (int s = 0; s < NSTRIP; ++s)
#pragma unroll
    for (int t = 0; t < NTILE; ++t)
#pragma unroll
      for (int r = 0; r < 8; ++r)
        wbase[(s * 16 + half * 8 + r) * NCOL + t * 16 + laneM] = acc[s][t][r];

  __syncthreads();

  // ---- Epilogue: one thread per row (64 threads, 64 rows)
  {
    const float* rv = smem + tid * NCOL;
    const int row = blockIdx.x * ROWS_PER_BLOCK + tid;
    float* orow = out + (size_t)row * NOUT;
#pragma unroll
    for (int t = 0; t < NTOP; ++t) {
      float ts = 1.0f / (1.0f + __expf(-(rv[t] + top_b[t])));
      float l[NCLS];
      float m = -3.4e38f;
#pragma unroll
      for (int c = 0; c < NCLS; ++c) {
        l[c] = rv[NTOP + t * NCLS + c] + bot_b[t * NCLS + c];
        m = fmaxf(m, l[c]);
      }
      float s = 0.0f;
#pragma unroll
      for (int c = 0; c < NCLS; ++c) { l[c] = __expf(l[c] - m); s += l[c]; }
      float inv = ts / s;
#pragma unroll
      for (int c4 = 0; c4 < NCLS; c4 += 4) {
        float4 o = make_float4(l[c4] * inv, l[c4 + 1] * inv,
                               l[c4 + 2] * inv, l[c4 + 3] * inv);
        *(float4*)(orow + t * NCLS + c4) = o;
      }
    }
  }
}

// ---------------------------------------------------------------------------
// Launcher
// inputs: 0=features (B,2048) f32, 1=top_W (8,2048) f32, 2=top_b (8,) f32,
//         3=bottom_W (8,2048,16) f32, 4=bottom_b (8,16) f32
// output: (B,128) f32
// ---------------------------------------------------------------------------
extern "C" void kernel_launch(void* const* d_in, const int* in_sizes, int n_in,
                              void* d_out, int out_size, void* d_ws, size_t ws_size,
                              hipStream_t stream) {
  const float* feat = (const float*)d_in[0];
  const float* topW = (const float*)d_in[1];
  const float* topb = (const float*)d_in[2];
  const float* botW = (const float*)d_in[3];
  const float* botb = (const float*)d_in[4];
  float* out = (float*)d_out;

  const int nrows = in_sizes[0] / DIMD;  // 16384

  // workspace: bf16 hi plane then bf16 lo plane, each 144*2048 elements
  __bf16* whi = (__bf16*)d_ws;
  __bf16* wlo = (__bf16*)((char*)d_ws + (size_t)NCOL * DIMD * sizeof(__bf16));

  const int prepN = NCOL * DIMD;
  hc_prep_weights<<<(prepN + 255) / 256, 256, 0, stream>>>(topW, botW, whi, wlo);

  const int nblocks = nrows / ROWS_PER_BLOCK;
  hc_main<<<nblocks, THREADS, 0, stream>>>(feat, whi, wlo, topb, botb, out);
}
